// SelfSelfAttention_25520695672850
// MI455X (gfx1250) — matrix operand order repaired and checked by reading, not compiled, not run on hardware
//
#include <hip/hip_runtime.h>
#include <hip/hip_bf16.h>
#include <stdint.h>

// ---------------------------------------------------------------------------
// SelfSelfAttention for MI455X (gfx1250, wave32, WMMA + async-to-LDS).
// B=16 H=16 N=577 D=64 C=1024.  ~250 GFLOP vs ~130 MB traffic -> compute
// bound -> all GEMM-shaped math via v_wmma_f32_16x16x32_bf16, fp32 accum.
// Global->LDS movement uses GLOBAL_LOAD_ASYNC_TO_LDS_B128 with double
// buffering (ASYNCcnt), so WMMA issue overlaps the next tile's fetch.
// ---------------------------------------------------------------------------

typedef __attribute__((ext_vector_type(16))) __bf16 v16bf;
typedef __attribute__((ext_vector_type(8)))  __bf16 v8bf;
typedef __attribute__((ext_vector_type(8)))  float  v8f;

#define NTOK 577
#define NB   16
#define NH   16
#define ND   64
#define NC   1024

static __device__ __forceinline__ v8f wmma_bf16(v16bf a, v16bf b, v8f c) {
  return __builtin_amdgcn_wmma_f32_16x16x32_bf16(false, a, false, b, (short)0, c,
                                                 false, false);
}

// concat two contiguous 16B chunks into one A/B fragment register block
static __device__ __forceinline__ v16bf cat8(v8bf lo, v8bf hi) {
  v16bf r;
#pragma unroll
  for (int i = 0; i < 8; ++i) {
    r[i] = lo[i];
    r[8 + i] = hi[i];
  }
  return r;
}

// ---- gfx1250 async global->LDS copies (ISA 15.18.3 op 98) -----------------
#define WAIT_ASYNC(n) asm volatile("s_wait_asynccnt " #n ::: "memory")

#define ASYNC_COPY_16B(lds, g)                                        \
  asm volatile("global_load_async_to_lds_b128 %0, %1, off" ::"v"(lds), \
               "v"(g)                                                 \
               : "memory")

#define ASYNC_COPY_32B(lds, g)                                                 \
  asm volatile(                                                                \
      "global_load_async_to_lds_b128 %0, %1, off\n\t"                          \
      "global_load_async_to_lds_b128 %0, %1, off offset:16" ::"v"(lds),        \
      "v"(g)                                                                   \
      : "memory")

#define ASYNC_COPY_128B(lds, g)                                                \
  asm volatile(                                                                \
      "global_load_async_to_lds_b128 %0, %1, off\n\t"                          \
      "global_load_async_to_lds_b128 %0, %1, off offset:16\n\t"                \
      "global_load_async_to_lds_b128 %0, %1, off offset:32\n\t"                \
      "global_load_async_to_lds_b128 %0, %1, off offset:48\n\t"                \
      "global_load_async_to_lds_b128 %0, %1, off offset:64\n\t"                \
      "global_load_async_to_lds_b128 %0, %1, off offset:80\n\t"                \
      "global_load_async_to_lds_b128 %0, %1, off offset:96\n\t"                \
      "global_load_async_to_lds_b128 %0, %1, off offset:112" ::"v"(lds),       \
      "v"(g)                                                                   \
      : "memory")

static __device__ __forceinline__ unsigned lds_off(const void* p) {
  return (unsigned)(uintptr_t)p;  // LDS offset = low 32 bits of generic addr
}
static __device__ __forceinline__ unsigned long long gaddr(const void* p) {
  return (unsigned long long)(uintptr_t)p;
}

// --------------------------- prep: transpose + row norms -------------------
__global__ __launch_bounds__(256) void prep_kernel(const float* __restrict__ x,
                                                   __bf16* __restrict__ xb,
                                                   float* __restrict__ rown) {
  int bn = blockIdx.x;  // bn = b*NTOK + n
  int b = bn / NTOK, n = bn % NTOK;
  const float* src = x + ((size_t)n * NB + b) * NC;  // x is [N,B,C]
  __bf16* dst = xb + (size_t)bn * NC;
  float ss = 0.f;
  for (int c = threadIdx.x; c < NC; c += 256) {
    float v = src[c];
    dst[c] = (__bf16)v;
    ss += v * v;
  }
  __shared__ float red[256];
  red[threadIdx.x] = ss;
  __syncthreads();
  for (int s = 128; s > 0; s >>= 1) {
    if (threadIdx.x < s) red[threadIdx.x] += red[threadIdx.x + s];
    __syncthreads();
  }
  if (threadIdx.x == 0) rown[bn] = sqrtf(red[0]);
}

__global__ __launch_bounds__(256) void invtemp_kernel(
    const float* __restrict__ rown, float* __restrict__ invt) {
  int b = blockIdx.x;
  __shared__ float red[256];
  float s = 0.f;
  for (int i = threadIdx.x; i < NTOK; i += 256) s += rown[b * NTOK + i];
  red[threadIdx.x] = s;
  __syncthreads();
  for (int st = 128; st > 0; st >>= 1) {
    if (threadIdx.x < st) red[threadIdx.x] += red[threadIdx.x + st];
    __syncthreads();
  }
  if (threadIdx.x == 0) invt[b] = (red[0] / (float)NTOK) * 0.125f;  // D^-0.5
}

__global__ __launch_bounds__(256) void wconv_kernel(const float* __restrict__ w,
                                                    __bf16* __restrict__ o,
                                                    long n) {
  long i = (long)blockIdx.x * 256 + threadIdx.x;
  if (i < n) o[i] = (__bf16)w[i];
}

// --------------------------- tiled bf16 WMMA GEMM --------------------------
// C[M,Ncols] = A[M,K] * W[Ncols,K]^T.  128x64 tile / WG, 8 waves (16x64 each),
// K stepped by 32 with double-buffered async global->LDS staging.
// mode 0: scatter into q/k/v [B,H,N,D] bf16
// mode 1: fp32 + bias, scattered into out laid out as [N,B,Ncols]
__global__ __launch_bounds__(256) void gemm_kernel(
    const __bf16* __restrict__ A, const __bf16* __restrict__ W, int M, int K,
    int Ncols, int mode, __bf16* __restrict__ qo, __bf16* __restrict__ ko,
    __bf16* __restrict__ vo, float* __restrict__ out,
    const float* __restrict__ bias) {
  __shared__ __align__(16) __bf16 As[2][128][40];  // 80B row stride
  __shared__ __align__(16) __bf16 Bs[2][64][40];

  const int tid = threadIdx.x;
  const int wave = tid >> 5, lane = tid & 31;
  const int grp = lane >> 4, ln = lane & 15;
  const int mBase = blockIdx.y * 128, nBase = blockIdx.x * 64;

  // loader mapping: A tile 128x32 (32B/thread), B tile 64x32 (16B/thread)
  const int tra = tid >> 1, halfa = tid & 1;
  const int trb = tid >> 2, qb = tid & 3;
  const int arow = (mBase + tra < M) ? mBase + tra : M - 1;
  const int brow = (nBase + trb < Ncols) ? nBase + trb : Ncols - 1;
  const __bf16* gA = A + (size_t)arow * K + halfa * 16;
  const __bf16* gB = W + (size_t)brow * K + qb * 8;
  const unsigned ldsA = lds_off(&As[0][tra][halfa * 16]);
  const unsigned ldsB = lds_off(&Bs[0][trb][qb * 8]);
  const unsigned bufA = sizeof(As) / 2;
  const unsigned bufB = sizeof(Bs) / 2;

  v8f cacc[4];
#pragma unroll
  for (int i = 0; i < 4; ++i) cacc[i] = (v8f){};

  const int KS = K >> 5;
  ASYNC_COPY_32B(ldsA, gaddr(gA));  // prologue: stage k-step 0 into buffer 0
  ASYNC_COPY_16B(ldsB, gaddr(gB));

  for (int it = 0; it < KS; ++it) {
    const int cur = it & 1;
    if (it + 1 < KS) {  // prefetch next k-step into the other buffer
      const unsigned nb_ = (unsigned)(cur ^ 1);
      ASYNC_COPY_32B(ldsA + nb_ * bufA, gaddr(gA + (it + 1) * 32));
      ASYNC_COPY_16B(ldsB + nb_ * bufB, gaddr(gB + (it + 1) * 32));
      WAIT_ASYNC(3);  // keep only the 3 just-issued prefetches in flight
    } else {
      WAIT_ASYNC(0);
    }
    __syncthreads();

    const int am = wave * 16 + ln;
    v16bf af = cat8(*(const v8bf*)&As[cur][am][grp * 16],
                    *(const v8bf*)&As[cur][am][grp * 16 + 8]);
#pragma unroll
    for (int nb = 0; nb < 4; ++nb) {
      v16bf bfb = cat8(*(const v8bf*)&Bs[cur][nb * 16 + ln][grp * 16],
                       *(const v8bf*)&Bs[cur][nb * 16 + ln][grp * 16 + 8]);
      cacc[nb] = wmma_bf16(af, bfb, cacc[nb]);
    }
    __syncthreads();
  }

  // epilogue: C element (m,n): VGPR r -> m = r + 8*grp, n = ln
#pragma unroll
  for (int nb = 0; nb < 4; ++nb) {
#pragma unroll
    for (int r = 0; r < 8; ++r) {
      int row = mBase + wave * 16 + r + 8 * grp;
      int col = nBase + nb * 16 + ln;
      if (row < M && col < Ncols) {
        float val = cacc[nb][r];
        int b = row / NTOK, n = row % NTOK;
        if (mode == 0) {
          int i = col >> 10, rem = col & 1023;
          int h = rem >> 6, d = rem & 63;
          __bf16* dst = (i == 0) ? qo : (i == 1) ? ko : vo;
          dst[(((size_t)b * NH + h) * NTOK + n) * ND + d] = (__bf16)val;
        } else {
          out[(size_t)n * (NB * Ncols) + (size_t)b * Ncols + col] =
              val + bias[col];
        }
      }
    }
  }
}

// --------------------------- flash attention -------------------------------
// One wave per (b,h, 16-row block).  K/V tiles (32 keys) double-buffered in
// LDS via async loads; S = Q K^T (4 WMMA), online softmax, P through LDS ->
// A-fragment, O += P V (4 WMMA).  Inputs [B,H,N,D] bf16, indices clamped
// (mask handles validity), so every load is unconditional.
__global__ __launch_bounds__(32) void attn_kernel(
    const __bf16* __restrict__ Qp, const __bf16* __restrict__ Kp,
    const __bf16* __restrict__ Vp, const float* __restrict__ invt, float cscale,
    int use_invt, __bf16* __restrict__ Op, int out_bnhd) {
  __shared__ __align__(16) __bf16 Ks[2][32][80];  // 160B row stride
  __shared__ __align__(16) __bf16 Vs[2][32][80];
  __shared__ __align__(16) __bf16 plds[16][32];

  const int lane = threadIdx.x, grp = lane >> 4, ln = lane & 15;
  const int bh = blockIdx.y, b = bh >> 4, h = bh & 15;
  const int m0 = blockIdx.x * 16;
  const float temp = use_invt ? invt[b] : cscale;

  const __bf16* Qh = Qp + (size_t)bh * NTOK * ND;
  const __bf16* Kh = Kp + (size_t)bh * NTOK * ND;
  const __bf16* Vh = Vp + (size_t)bh * NTOK * ND;

  // Q A-fragments (d 0..31 and 32..63); rows clamped, invalid rows unstored
  const int qrow = (m0 + ln < NTOK) ? m0 + ln : NTOK - 1;
  const __bf16* qp = Qh + (size_t)qrow * ND;
  v16bf qa0 = cat8(*(const v8bf*)(qp + grp * 16), *(const v8bf*)(qp + grp * 16 + 8));
  v16bf qa1 =
      cat8(*(const v8bf*)(qp + 32 + grp * 16), *(const v8bf*)(qp + 40 + grp * 16));

  v8f acc[4];
#pragma unroll
  for (int i = 0; i < 4; ++i) acc[i] = (v8f){};
  float mrun[8], lrun[8];
#pragma unroll
  for (int r = 0; r < 8; ++r) {
    mrun[r] = -1e30f;
    lrun[r] = 0.f;
  }

  const unsigned ldsK = lds_off(&Ks[0][lane][0]);
  const unsigned ldsV = lds_off(&Vs[0][lane][0]);
  const unsigned bufKV = sizeof(Ks) / 2;

  // prologue: stage keys [0,32) into buffer 0 (one 128B row per lane)
  {
    const int key = (lane < NTOK) ? lane : NTOK - 1;
    ASYNC_COPY_128B(ldsK, gaddr(Kh + (size_t)key * ND));
    ASYNC_COPY_128B(ldsV, gaddr(Vh + (size_t)key * ND));
  }

  const int NIT = (NTOK + 31) / 32;  // 19
  for (int it = 0; it < NIT; ++it) {
    const int j0 = it * 32, cur = it & 1;
    if (it + 1 < NIT) {
      const int key = (j0 + 32 + lane < NTOK) ? j0 + 32 + lane : NTOK - 1;
      const unsigned nb_ = (unsigned)(cur ^ 1);
      ASYNC_COPY_128B(ldsK + nb_ * bufKV, gaddr(Kh + (size_t)key * ND));
      ASYNC_COPY_128B(ldsV + nb_ * bufKV, gaddr(Vh + (size_t)key * ND));
      WAIT_ASYNC(16);  // current tile complete; only prefetch in flight
    } else {
      WAIT_ASYNC(0);
    }

    // K B-fragments: 2 key sub-blocks x 2 d-halves
    v16bf ka0 = cat8(*(const v8bf*)&Ks[cur][ln][grp * 16],
                     *(const v8bf*)&Ks[cur][ln][grp * 16 + 8]);
    v16bf ka1 = cat8(*(const v8bf*)&Ks[cur][ln][32 + grp * 16],
                     *(const v8bf*)&Ks[cur][ln][32 + grp * 16 + 8]);
    v16bf kb0 = cat8(*(const v8bf*)&Ks[cur][16 + ln][grp * 16],
                     *(const v8bf*)&Ks[cur][16 + ln][grp * 16 + 8]);
    v16bf kb1 = cat8(*(const v8bf*)&Ks[cur][16 + ln][32 + grp * 16],
                     *(const v8bf*)&Ks[cur][16 + ln][32 + grp * 16 + 8]);

    v8f s0 = {}, s1 = {};
    s0 = wmma_bf16(qa0, ka0, s0);
    s0 = wmma_bf16(qa1, ka1, s0);
    s1 = wmma_bf16(qa0, kb0, s1);
    s1 = wmma_bf16(qa1, kb1, s1);

    const bool va = (j0 + ln) < NTOK;
    const bool vb = (j0 + 16 + ln) < NTOK;

    // online softmax; row = r + 8*grp, col = ln
#pragma unroll
    for (int r = 0; r < 8; ++r) {
      float x0 = va ? s0[r] * temp : -1e30f;
      float x1 = vb ? s1[r] * temp : -1e30f;
      float mx = fmaxf(x0, x1);
#pragma unroll
      for (int off = 8; off >= 1; off >>= 1)
        mx = fmaxf(mx, __shfl_xor(mx, off, 32));
      float mnew = fmaxf(mrun[r], mx);
      float corr = __expf(mrun[r] - mnew);
      float p0 = __expf(x0 - mnew);
      float p1 = __expf(x1 - mnew);
      float rs = p0 + p1;
#pragma unroll
      for (int off = 8; off >= 1; off >>= 1) rs += __shfl_xor(rs, off, 32);
      lrun[r] = lrun[r] * corr + rs;
      mrun[r] = mnew;
      acc[0][r] *= corr;
      acc[1][r] *= corr;
      acc[2][r] *= corr;
      acc[3][r] *= corr;
      plds[r + 8 * grp][ln] = (__bf16)p0;
      plds[r + 8 * grp][16 + ln] = (__bf16)p1;
    }
    __syncthreads();

    // P A-fragment (16 rows x 32 keys)
    v16bf pa = cat8(*(const v8bf*)&plds[ln][grp * 16],
                    *(const v8bf*)&plds[ln][grp * 16 + 8]);

    // O += P * V  (V B-fragment: k = key offset, n = d) from staged tile
#pragma unroll
    for (int db = 0; db < 4; ++db) {
      v16bf vf;
      const int d = db * 16 + ln;
#pragma unroll
      for (int i = 0; i < 16; ++i) vf[i] = Vs[cur][grp * 16 + i][d];
      acc[db] = wmma_bf16(pa, vf, acc[db]);
    }
    __syncthreads();
  }

  // store: [B,H,N,D] (chaining) or [B,N,H,D]=[B,N,C] (proj-ready)
#pragma unroll
  for (int db = 0; db < 4; ++db) {
#pragma unroll
    for (int r = 0; r < 8; ++r) {
      int row = m0 + r + 8 * grp;
      if (row < NTOK) {
        float val = acc[db][r] / lrun[r];
        int d = db * 16 + ln;
        size_t idx = out_bnhd ? ((((size_t)b * NTOK + row) * NH + h) * ND + d)
                              : (((size_t)bh * NTOK + row) * ND + d);
        Op[idx] = (__bf16)val;
      }
    }
  }
}

// --------------------------- L2 normalize rows of 64 -----------------------
__global__ __launch_bounds__(128) void norm_kernel(const __bf16* __restrict__ in,
                                                   __bf16* __restrict__ out,
                                                   long nrows) {
  long row = (long)blockIdx.x * 4 + (threadIdx.x >> 5);
  int l = threadIdx.x & 31;
  if (row >= nrows) return;
  const __bf16* p = in + row * ND;
  float a = (float)p[l], b = (float)p[l + 32];
  float ss = a * a + b * b;
#pragma unroll
  for (int off = 16; off >= 1; off >>= 1) ss += __shfl_xor(ss, off, 32);
  float inv = 1.0f / fmaxf(sqrtf(ss), 1e-12f);
  out[row * ND + l] = (__bf16)(a * inv);
  out[row * ND + l + 32] = (__bf16)(b * inv);
}

// --------------------------- 3-way average ---------------------------------
__global__ __launch_bounds__(256) void combine_kernel(
    const __bf16* __restrict__ a, const __bf16* __restrict__ b,
    const __bf16* __restrict__ c, __bf16* __restrict__ o, long n) {
  long i = (long)blockIdx.x * 256 + threadIdx.x;
  if (i < n)
    o[i] = (__bf16)(((float)a[i] + (float)b[i] + (float)c[i]) * (1.0f / 3.0f));
}

// ---------------------------------------------------------------------------
extern "C" void kernel_launch(void* const* d_in, const int* in_sizes, int n_in,
                              void* d_out, int out_size, void* d_ws,
                              size_t ws_size, hipStream_t stream) {
  (void)in_sizes; (void)n_in; (void)out_size; (void)ws_size;
  const float* x = (const float*)d_in[0];       // [N,B,C]
  const float* qkv_w = (const float*)d_in[1];   // [3C,C]
  const float* proj_w = (const float*)d_in[2];  // [C,C]
  const float* proj_b = (const float*)d_in[3];  // [C]
  float* out = (float*)d_out;                   // [2, N, B, C]

  // workspace layout (~198 MB)
  const size_t SZ = (size_t)NB * NH * NTOK * ND * 2;  // 18,907,136 B
  char* w = (char*)d_ws;
  __bf16* XB   = (__bf16*)(w + 0 * SZ);  // [B*N, C]
  __bf16* Q    = (__bf16*)(w + 1 * SZ);  // [B,H,N,D]
  __bf16* K    = (__bf16*)(w + 2 * SZ);
  __bf16* V    = (__bf16*)(w + 3 * SZ);
  __bf16* TNV  = (__bf16*)(w + 4 * SZ);  // norm(v), later s2_v [B,N,C]
  __bf16* TNK  = (__bf16*)(w + 5 * SZ);  // norm(k), later s2_k
  __bf16* TNQ  = (__bf16*)(w + 6 * SZ);  // norm(q), later s2_q
  __bf16* T1   = (__bf16*)(w + 7 * SZ);  // s1_v (then normalized in place)
  __bf16* XORI = (__bf16*)(w + 8 * SZ);  // x_ori heads, [B,N,C]
  __bf16* XS   = (__bf16*)(w + 9 * SZ);  // combined streams, [B,N,C]
  __bf16* WQKV = (__bf16*)(w + 10 * SZ);
  __bf16* WPRJ = (__bf16*)(w + 10 * SZ + (size_t)3 * NC * NC * 2);
  float* ROWN  = (float*)(w + 10 * SZ + (size_t)4 * NC * NC * 2);
  float* INVT  = ROWN + NB * NTOK;

  const int M = NB * NTOK;  // 9232
  const long NROWS = (long)NB * NH * NTOK;
  const long NEL = (long)NB * NTOK * NC;
  const size_t NBC = (size_t)NTOK * NB * NC;

  // 0) weights -> bf16
  wconv_kernel<<<(3 * NC * NC + 255) / 256, 256, 0, stream>>>(qkv_w, WQKV,
                                                              3L * NC * NC);
  wconv_kernel<<<(NC * NC + 255) / 256, 256, 0, stream>>>(proj_w, WPRJ,
                                                          (long)NC * NC);
  // 1) transpose + row norms, inv temperature
  prep_kernel<<<M, 256, 0, stream>>>(x, XB, ROWN);
  invtemp_kernel<<<NB, 256, 0, stream>>>(ROWN, INVT);

  // 2) QKV GEMM: [9232,1024] x [3072,1024]^T, scatter into Q/K/V
  gemm_kernel<<<dim3((3 * NC) / 64, (M + 127) / 128), 256, 0, stream>>>(
      XB, WQKV, M, NC, 3 * NC, 0, Q, K, V, nullptr, nullptr);

  dim3 agrid((NTOK + 15) / 16, NB * NH);
  // 3) standard attention branch -> XORI in [B,N,C]
  attn_kernel<<<agrid, 32, 0, stream>>>(Q, K, V, INVT, 0.125f, 0, XORI, 1);

  // 4) normalize q,k,v
  norm_kernel<<<(NROWS + 3) / 4, 128, 0, stream>>>(V, TNV, NROWS);
  norm_kernel<<<(NROWS + 3) / 4, 128, 0, stream>>>(K, TNK, NROWS);
  norm_kernel<<<(NROWS + 3) / 4, 128, 0, stream>>>(Q, TNQ, NROWS);

  // 5) first ss pass: s = softmax(n(s) n(s)^T * invT) n(s)
  attn_kernel<<<agrid, 32, 0, stream>>>(TNV, TNV, TNV, INVT, 0.f, 1, T1, 0);
  attn_kernel<<<agrid, 32, 0, stream>>>(TNK, TNK, TNK, INVT, 0.f, 1, Q, 0);
  attn_kernel<<<agrid, 32, 0, stream>>>(TNQ, TNQ, TNQ, INVT, 0.f, 1, K, 0);

  // 6) normalize pass-1 streams in place
  norm_kernel<<<(NROWS + 3) / 4, 128, 0, stream>>>(T1, T1, NROWS);
  norm_kernel<<<(NROWS + 3) / 4, 128, 0, stream>>>(Q, Q, NROWS);
  norm_kernel<<<(NROWS + 3) / 4, 128, 0, stream>>>(K, K, NROWS);

  // 7) second ss pass with value = v, outputs in [B,N,C]
  attn_kernel<<<agrid, 32, 0, stream>>>(T1, T1, V, INVT, 0.f, 1, TNV, 1);
  attn_kernel<<<agrid, 32, 0, stream>>>(Q, Q, V, INVT, 0.f, 1, TNK, 1);
  attn_kernel<<<agrid, 32, 0, stream>>>(K, K, V, INVT, 0.f, 1, TNQ, 1);

  // 8) average the three streams
  combine_kernel<<<(NEL + 255) / 256, 256, 0, stream>>>(TNV, TNK, TNQ, XS, NEL);

  // 9) output projections (epilogue writes [N,B,C] + bias directly to d_out)
  gemm_kernel<<<dim3(NC / 64, (M + 127) / 128), 256, 0, stream>>>(
      XS, WPRJ, M, NC, NC, 1, nullptr, nullptr, nullptr, out, proj_b);
  gemm_kernel<<<dim3(NC / 64, (M + 127) / 128), 256, 0, stream>>>(
      XORI, WPRJ, M, NC, NC, 1, nullptr, nullptr, nullptr, out + NBC, proj_b);
}
